// CCRNN_70231305224795
// MI455X (gfx1250) — compile-verified, matrix operand order checked by reading, not run on hardware
//
#include <hip/hip_runtime.h>
#include <hip/hip_bf16.h>

// ---------------------------------------------------------------------------
// CCRNN on MI455X (gfx1250): bf16 WMMA GEMMs + fully fused LSTM recurrence.
// ---------------------------------------------------------------------------

#define BB   128
#define TT   40
#define EE   512
#define HH   1024
#define VV   10000
#define ININ 2048

typedef __attribute__((ext_vector_type(16))) __bf16 v16bf;
typedef __attribute__((ext_vector_type(8)))  __bf16 v8bf;
typedef __attribute__((ext_vector_type(8)))  float  v8f;

union FragAB { v16bf v; v8bf h[2]; };

__device__ __forceinline__ unsigned short f2bf_u16(float f) {
    union { float f; unsigned u; } x; x.f = f;
    unsigned r = x.u + 0x7fffu + ((x.u >> 16) & 1u);   // round-to-nearest-even
    return (unsigned short)(r >> 16);
}

__device__ __forceinline__ float sigmoidf_fast(float x) {
    return 1.0f / (1.0f + __expf(-x));
}

// ---------------------------------------------------------------------------
// fp32 -> bf16 conversion
// ---------------------------------------------------------------------------
__global__ void k_convert_bf16(const float* __restrict__ in,
                               unsigned short* __restrict__ out, int n) {
    int i = blockIdx.x * blockDim.x + threadIdx.x;
    if (i < n) out[i] = f2bf_u16(in[i]);
}

// ---------------------------------------------------------------------------
// Build x_seq bf16 [T*B, 2E]: row m = t*B + b ; first E = feats_bf[b],
// second E = bf16(lembed[labels[b][t]])
// ---------------------------------------------------------------------------
__global__ void k_build_xseq(const unsigned short* __restrict__ featsBf,
                             const float* __restrict__ lembed,
                             const int* __restrict__ labels,
                             unsigned short* __restrict__ xseq, int n_total) {
    int idx = blockIdx.x * blockDim.x + threadIdx.x;
    if (idx >= n_total) return;
    const int twoE = 2 * EE;
    int m = idx / twoE, e = idx % twoE;
    int t = m / BB, b = m % BB;
    unsigned short v;
    if (e < EE) {
        v = featsBf[b * EE + e];
    } else {
        int lab = labels[b * TT + t];
        v = f2bf_u16(lembed[(size_t)lab * EE + (e - EE)]);
    }
    xseq[idx] = v;
}

// ---------------------------------------------------------------------------
// Generic bf16 WMMA GEMM:  C[M,N] = A[M,K] * W[N,K]^T (+ bias)
// Wave tile 64x32 = 4x2 WMMA 16x16 tiles; 8 waves per 256-thread block.
// Fragment layout per CDNA5 ISA 7.12.2 (16-bit A 16x32; B mirrored; C 8 VGPR).
// Options: bf16 output copy, (t*B+b -> b*T+t) row remap, N edge guard,
// nontemporal stores (TH=NT) for the streaming 205MB logits output.
// ---------------------------------------------------------------------------
__global__ __launch_bounds__(256)
void k_gemm_bf16(const unsigned short* __restrict__ A, int lda,
                 const unsigned short* __restrict__ W, int ldw,
                 const float* __restrict__ bias,
                 float* __restrict__ Cout,
                 unsigned short* __restrict__ CoutBf, int ldc,
                 int M, int Nreal, int Ntiles32, int K,
                 int remapB, int remapT, int ntOut) {
    const int wave = (blockIdx.x * blockDim.x + threadIdx.x) >> 5;
    const int lane = threadIdx.x & 31;
    const int numTiles = (M >> 6) * Ntiles32;
    if (wave >= numTiles) return;

    const int tm = wave / Ntiles32;
    const int tn = wave % Ntiles32;
    const int mbase = tm << 6;            // 64-row tile
    const int nbase = tn << 5;            // 32-col tile
    const int half  = lane >> 4;          // K-half select per ISA layout
    const int lan   = lane & 15;
    const int koff  = half << 3;          // lanes 0-15: K{0..7,16..23}; 16-31: +8

    v8f acc[4][2];
#pragma unroll
    for (int i = 0; i < 4; ++i)
#pragma unroll
        for (int j = 0; j < 2; ++j)
#pragma unroll
            for (int r = 0; r < 8; ++r) acc[i][j][r] = 0.0f;

    int nrow[2];                          // clamped W rows (V=10000 edge)
#pragma unroll
    for (int j = 0; j < 2; ++j) {
        int n = nbase + j * 16 + lan;
        nrow[j] = (n < Nreal) ? n : (Nreal - 1);
    }

    for (int kk = 0; kk < K; kk += 32) {
        FragAB a[4], b[2];
#pragma unroll
        for (int i = 0; i < 4; ++i) {
            const unsigned short* pa =
                A + (size_t)(mbase + i * 16 + lan) * lda + kk + koff;
            a[i].h[0] = *reinterpret_cast<const v8bf*>(pa);        // K 0..7
            a[i].h[1] = *reinterpret_cast<const v8bf*>(pa + 16);   // K 16..23
        }
#pragma unroll
        for (int j = 0; j < 2; ++j) {
            const unsigned short* pw = W + (size_t)nrow[j] * ldw + kk + koff;
            b[j].h[0] = *reinterpret_cast<const v8bf*>(pw);
            b[j].h[1] = *reinterpret_cast<const v8bf*>(pw + 16);
        }
#pragma unroll
        for (int i = 0; i < 4; ++i)
#pragma unroll
            for (int j = 0; j < 2; ++j)
                acc[i][j] = __builtin_amdgcn_wmma_f32_16x16x32_bf16(
                    false, a[i].v, false, b[j].v, (short)0, acc[i][j],
                    false, false);
    }

#pragma unroll
    for (int j = 0; j < 2; ++j) {
        int col = nbase + j * 16 + lan;
        bool colOK = (col < Nreal);
        float bv = bias ? bias[colOK ? col : 0] : 0.0f;
#pragma unroll
        for (int i = 0; i < 4; ++i) {
#pragma unroll
            for (int r = 0; r < 8; ++r) {
                int row = mbase + i * 16 + half * 8 + r;
                if (colOK && row < M) {
                    float v = acc[i][j][r] + bv;
                    size_t orow = (size_t)row;
                    if (remapB > 0) {
                        int t = row / remapB, bb = row % remapB;
                        orow = (size_t)bb * remapT + t;
                    }
                    size_t off2 = orow * (size_t)ldc + col;
                    if (Cout) {
                        if (ntOut) __builtin_nontemporal_store(v, &Cout[off2]);
                        else       Cout[off2] = v;
                    }
                    if (CoutBf) CoutBf[off2] = f2bf_u16(v);
                }
            }
        }
    }
}

// ---------------------------------------------------------------------------
// Fully fused LSTM step:  gates = h@w_hh^T + b_hh + gx_t, then the cell
// update, in one kernel.  Each wave owns a 16-row x 32-col tile of the
// hidden state and accumulates ALL FOUR gate blocks for that tile
// (acc[g][j], 8 WMMA tiles = 64 VGPRs), so sigmoid/tanh/c/h happen
// in-register: no gates buffer, one launch per time step.
// ---------------------------------------------------------------------------
__global__ __launch_bounds__(256)
void k_lstm_step_fused(const unsigned short* __restrict__ hIn,   // [B,H] bf16
                       const unsigned short* __restrict__ Whh,   // [4H,H] bf16
                       const float* __restrict__ bhh,            // [4H]
                       const float* __restrict__ gxT,            // [B,4H] f32
                       float* __restrict__ c,                    // [B,H]
                       unsigned short* __restrict__ hOut,        // [B,H] bf16
                       unsigned short* __restrict__ hsT) {       // [B,H] bf16
    const int wave = (blockIdx.x * blockDim.x + threadIdx.x) >> 5;
    const int lane = threadIdx.x & 31;
    const int nTilesN = HH / 32;                   // 32
    const int numTiles = (BB / 16) * nTilesN;      // 256 waves total
    if (wave >= numTiles) return;

    const int mbase = (wave / nTilesN) << 4;       // 16-row tile of batch
    const int nbase = (wave % nTilesN) << 5;       // 32-col tile of hidden
    const int half  = lane >> 4;
    const int lan   = lane & 15;
    const int koff  = half << 3;

    v8f acc[4][2];
#pragma unroll
    for (int g = 0; g < 4; ++g)
#pragma unroll
        for (int j = 0; j < 2; ++j)
#pragma unroll
            for (int r = 0; r < 8; ++r) acc[g][j][r] = 0.0f;

    const size_t aRow = (size_t)(mbase + lan) * HH;
    for (int kk = 0; kk < HH; kk += 32) {
        FragAB a, b[4][2];
        const unsigned short* pa = hIn + aRow + kk + koff;
        a.h[0] = *reinterpret_cast<const v8bf*>(pa);
        a.h[1] = *reinterpret_cast<const v8bf*>(pa + 16);
#pragma unroll
        for (int g = 0; g < 4; ++g)
#pragma unroll
            for (int j = 0; j < 2; ++j) {
                const unsigned short* pw =
                    Whh + (size_t)(g * HH + nbase + j * 16 + lan) * HH + kk + koff;
                b[g][j].h[0] = *reinterpret_cast<const v8bf*>(pw);
                b[g][j].h[1] = *reinterpret_cast<const v8bf*>(pw + 16);
            }
#pragma unroll
        for (int g = 0; g < 4; ++g)
#pragma unroll
            for (int j = 0; j < 2; ++j)
                acc[g][j] = __builtin_amdgcn_wmma_f32_16x16x32_bf16(
                    false, a.v, false, b[g][j].v, (short)0, acc[g][j],
                    false, false);
    }

    // In-register cell update for the 16x32 tile this wave owns.
#pragma unroll
    for (int j = 0; j < 2; ++j) {
        int col = nbase + j * 16 + lan;
#pragma unroll
        for (int r = 0; r < 8; ++r) {
            int row = mbase + half * 8 + r;
            const float* gxr = gxT + (size_t)row * 4 * HH;
            float gi = acc[0][j][r] + bhh[col]          + gxr[col];
            float gf = acc[1][j][r] + bhh[HH + col]     + gxr[HH + col];
            float gg = acc[2][j][r] + bhh[2 * HH + col] + gxr[2 * HH + col];
            float go = acc[3][j][r] + bhh[3 * HH + col] + gxr[3 * HH + col];
            float iv = sigmoidf_fast(gi);
            float fv = sigmoidf_fast(gf);
            float gv = tanhf(gg);
            float ov = sigmoidf_fast(go);
            size_t idx = (size_t)row * HH + col;
            float cn = fv * c[idx] + iv * gv;
            c[idx] = cn;
            unsigned short hb = f2bf_u16(ov * tanhf(cn));
            hOut[idx] = hb;
            hsT[idx]  = hb;
        }
    }
}

// ---------------------------------------------------------------------------
// Host-side orchestration
// ---------------------------------------------------------------------------
extern "C" void kernel_launch(void* const* d_in, const int* in_sizes, int n_in,
                              void* d_out, int out_size, void* d_ws, size_t ws_size,
                              hipStream_t stream) {
    (void)in_sizes; (void)n_in; (void)out_size; (void)ws_size;
    const float* X        = (const float*)d_in[0];
    const int*   labels   = (const int*)  d_in[1];
    const float* fembed_w = (const float*)d_in[2];
    const float* fembed_b = (const float*)d_in[3];
    const float* lembed   = (const float*)d_in[4];
    const float* w_ih     = (const float*)d_in[5];
    const float* b_ih     = (const float*)d_in[6];
    const float* w_hh     = (const float*)d_in[7];
    const float* b_hh     = (const float*)d_in[8];
    const float* lin_w    = (const float*)d_in[9];
    const float* lin_b    = (const float*)d_in[10];
    float* out = (float*)d_out;

    // Workspace carve-up (256B aligned)
    size_t off = 0;
    auto carve = [&](size_t bytes) -> size_t {
        size_t p = off; off = (off + bytes + 255) & ~(size_t)255; return p;
    };
    char* ws = (char*)d_ws;
    unsigned short* Xbf     = (unsigned short*)(ws + carve((size_t)BB * ININ * 2));
    unsigned short* Fwbf    = (unsigned short*)(ws + carve((size_t)EE * ININ * 2));
    unsigned short* featsBf = (unsigned short*)(ws + carve((size_t)BB * EE * 2));
    unsigned short* xseq    = (unsigned short*)(ws + carve((size_t)TT * BB * 2 * EE * 2));
    unsigned short* wihBf   = (unsigned short*)(ws + carve((size_t)4 * HH * 2 * EE * 2));
    unsigned short* whhBf   = (unsigned short*)(ws + carve((size_t)4 * HH * HH * 2));
    unsigned short* linwBf  = (unsigned short*)(ws + carve((size_t)VV * HH * 2));
    float*          gx      = (float*)         (ws + carve((size_t)TT * BB * 4 * HH * 4));
    float*          cbuf    = (float*)         (ws + carve((size_t)BB * HH * 4));
    unsigned short* hbf     = (unsigned short*)(ws + carve((size_t)BB * HH * 2));
    unsigned short* hsbf    = (unsigned short*)(ws + carve((size_t)TT * BB * HH * 2));

    const int THR = 256;
    auto cdiv = [](long long a, long long b) { return (int)((a + b - 1) / b); };

    // --- 1. Convert inputs/weights to bf16 (~37MB total; resident in 192MB L2)
    k_convert_bf16<<<cdiv((size_t)BB * ININ, THR), THR, 0, stream>>>(X, Xbf, BB * ININ);
    k_convert_bf16<<<cdiv((size_t)EE * ININ, THR), THR, 0, stream>>>(fembed_w, Fwbf, EE * ININ);
    k_convert_bf16<<<cdiv((size_t)4 * HH * 2 * EE, THR), THR, 0, stream>>>(w_ih, wihBf, 4 * HH * 2 * EE);
    k_convert_bf16<<<cdiv((size_t)4 * HH * HH, THR), THR, 0, stream>>>(w_hh, whhBf, 4 * HH * HH);
    k_convert_bf16<<<cdiv((size_t)VV * HH, THR), THR, 0, stream>>>(lin_w, linwBf, VV * HH);

    // Zero-init recurrent state (bf16 zero == 0x0000)
    hipMemsetAsync(cbuf, 0, (size_t)BB * HH * 4, stream);
    hipMemsetAsync(hbf,  0, (size_t)BB * HH * 2, stream);

    // --- 2. feats = X @ fembed_w^T + fembed_b  ->  bf16 [B,E] ---
    {
        int tiles = (BB / 64) * (EE / 32);
        k_gemm_bf16<<<cdiv(tiles, 8), THR, 0, stream>>>(
            Xbf, ININ, Fwbf, ININ, fembed_b,
            nullptr, featsBf, EE,
            BB, EE, EE / 32, ININ, 0, 0, 0);
    }

    // --- 3. x_seq bf16 [T*B, 2E] ---
    {
        int n = TT * BB * 2 * EE;
        k_build_xseq<<<cdiv(n, THR), THR, 0, stream>>>(featsBf, lembed, labels, xseq, n);
    }

    // --- 4. gx = x_seq @ w_ih^T + b_ih  -> f32 [T*B, 4H] ---
    {
        int M = TT * BB, N = 4 * HH;
        int tiles = (M / 64) * (N / 32);
        k_gemm_bf16<<<cdiv(tiles, 8), THR, 0, stream>>>(
            xseq, 2 * EE, wihBf, 2 * EE, b_ih,
            gx, nullptr, N,
            M, N, N / 32, 2 * EE, 0, 0, 0);
    }

    // --- 5. LSTM scan: 40 fully fused steps (GEMM + cell in one kernel) ---
    for (int t = 0; t < TT; ++t) {
        int tiles = (BB / 16) * (HH / 32);      // 256 waves
        k_lstm_step_fused<<<cdiv(tiles, 8), THR, 0, stream>>>(
            hbf, whhBf, b_hh,
            gx + (size_t)t * BB * 4 * HH,
            cbuf, hbf, hsbf + (size_t)t * BB * HH);
    }

    // --- 6. logits: out[b,t,v] = hs[t,b,:] . lin_w[v,:] + lin_b[v] ---
    //        nontemporal stores: 205MB output, written once, never re-read.
    {
        int M = TT * BB;
        int Ntiles = (VV + 31) / 32;            // 313 tiles, edge-guarded
        int tiles = (M / 64) * Ntiles;
        k_gemm_bf16<<<cdiv(tiles, 8), THR, 0, stream>>>(
            hsbf, HH, linwBf, HH, lin_b,
            out, nullptr, VV,
            M, VV, Ntiles, HH, BB, TT, 1);
    }
}
